// RecursiveBlock_49735721288462
// MI455X (gfx1250) — compile-verified, matrix operand order checked by reading, not run on hardware
//
#include <hip/hip_runtime.h>

// ---------------------------------------------------------------------------
// Types for CDNA5 WMMA (wave32, v_wmma_f32_16x16x32_bf16)
// ---------------------------------------------------------------------------
typedef __attribute__((ext_vector_type(16))) __bf16 bf16x16;
typedef __attribute__((ext_vector_type(8)))  float  f32x8;
typedef __attribute__((ext_vector_type(4)))  unsigned int uint4v;

union BFragU { bf16x16 v; uint4v q[2]; };

#define NROW 4096
#define CDIM 512
#define PL   256

// ---------------------------------------------------------------------------
// Fragment loaders
// A (16x32 bf16) from row-major bf16: lane l -> row (l&15), K-halves by g=l>>4:
//   elems 0..7  = K (8g .. 8g+7), elems 8..15 = K (16+8g .. 23+8g)
// ---------------------------------------------------------------------------
__device__ __forceinline__ bf16x16 load_afrag(const __bf16* rowptr, int colbase, int g) {
    BFragU f;
    f.q[0] = *reinterpret_cast<const uint4v*>(rowptr + colbase + 8 * g);
    f.q[1] = *reinterpret_cast<const uint4v*>(rowptr + colbase + 16 + 8 * g);
    return f.v;
}

// B fragments pre-packed lane-major: frag = 32 lanes x 16 bf16 (1 KB), fully
// coalesced: each lane loads 32 contiguous bytes (two b128 loads).
__device__ __forceinline__ bf16x16 load_bfrag(const __bf16* base, long fragIdx, int lane) {
    const __bf16* p = base + fragIdx * 512 + lane * 16;
    BFragU f;
    f.q[0] = *reinterpret_cast<const uint4v*>(p);
    f.q[1] = *reinterpret_cast<const uint4v*>(p + 8);
    return f.v;
}

__device__ __forceinline__ f32x8 wmma_bf16(bf16x16 a, bf16x16 b, f32x8 c) {
    return __builtin_amdgcn_wmma_f32_16x16x32_bf16(false, a, false, b, (short)0, c, false, false);
}

__device__ __forceinline__ float wave_sum(float v) {
    #pragma unroll
    for (int m = 16; m >= 1; m >>= 1) v += __shfl_xor(v, m, 32);
    return v;
}

// ---------------------------------------------------------------------------
// Pack: cat = [img || loc] -> bf16 row-major [4096][512]
// ---------------------------------------------------------------------------
__global__ void pack_cat_kernel(const float* __restrict__ img, const float* __restrict__ loc,
                                __bf16* __restrict__ dst) {
    int idx = blockIdx.x * 256 + threadIdx.x;          // [0, 4096*512)
    int row = idx >> 9, col = idx & 511;
    float v = (col < 256) ? img[row * 256 + col] : loc[row * 256 + (col - 256)];
    dst[idx] = (__bf16)v;
}

// ---------------------------------------------------------------------------
// Pack f32 [Kb x 256] (batched) -> bf16 fragment-linear [Kb/32][16][32][16]
// Inverse map of the WMMA 16-bit B layout (32x16 tile):
//   lane = n + 16g, elem e -> K = (e&7) + 16*(e>>3) + 8g
// ---------------------------------------------------------------------------
__global__ void pack_b_kernel(const float* __restrict__ src, __bf16* __restrict__ dst,
                              long perBatch, long srcStride, long dstStride) {
    long idx = (long)blockIdx.x * 256 + threadIdx.x;
    long b   = idx / perBatch;
    long d   = idx % perBatch;
    int e     = (int)(d & 15);
    int lane  = (int)((d >> 4) & 31);
    int ntile = (int)((d >> 9) & 15);
    int chunk = (int)(d >> 13);
    int g = lane >> 4, n = lane & 15;
    int kk = (e & 7) + ((e >> 3) << 4) + (g << 3);
    long k   = (long)chunk * 32 + kk;
    long col = (long)ntile * 16 + n;
    dst[b * dstStride + d] = (__bf16)src[b * srcStride + k * 256 + col];
}

// ---------------------------------------------------------------------------
// Generic bf16 GEMM (M x K x 256), fragment-packed B, optional ReLU,
// f32 or bf16 output.  WG = 8 waves arranged 4(M) x 2(N); tile 64x64.
// ---------------------------------------------------------------------------
__global__ __launch_bounds__(256) void gemm_bf16_kernel(
        const __bf16* __restrict__ A, int lda, const __bf16* __restrict__ Bp,
        const float* __restrict__ bias, void* __restrict__ out, int ldo,
        int ksteps, int outBf16, int doRelu) {
    int lane = threadIdx.x & 31, g = lane >> 4, ln15 = lane & 15;
    int wave = threadIdx.x >> 5;
    int r0 = blockIdx.x * 64 + (wave & 3) * 16;
    int c0 = blockIdx.y * 64 + (wave >> 2) * 32;
    int nt0 = c0 >> 4;
    const __bf16* arow = A + (long)(r0 + ln15) * lda;

    f32x8 acc0 = {0,0,0,0,0,0,0,0}, acc1 = {0,0,0,0,0,0,0,0};
    for (int ks = 0; ks < ksteps; ++ks) {
        bf16x16 a  = load_afrag(arow, ks * 32, g);
        bf16x16 b0 = load_bfrag(Bp, (long)ks * 16 + nt0,     lane);
        bf16x16 b1 = load_bfrag(Bp, (long)ks * 16 + nt0 + 1, lane);
        acc0 = wmma_bf16(a, b0, acc0);
        acc1 = wmma_bf16(a, b1, acc1);
    }
    float bia0 = bias[c0 + ln15], bia1 = bias[c0 + 16 + ln15];
    #pragma unroll
    for (int j = 0; j < 8; ++j) {
        int row = r0 + j + 8 * g;
        float f0 = acc0[j] + bia0, f1 = acc1[j] + bia1;
        if (doRelu) { f0 = fmaxf(f0, 0.f); f1 = fmaxf(f1, 0.f); }
        long i0 = (long)row * ldo + c0 + ln15;
        if (outBf16) {
            ((__bf16*)out)[i0]      = (__bf16)f0;
            ((__bf16*)out)[i0 + 16] = (__bf16)f1;
        } else {
            ((float*)out)[i0]      = f0;
            ((float*)out)[i0 + 16] = f1;
        }
    }
}

// ---------------------------------------------------------------------------
// Fused hypernetwork bilinear kernel:
//   out[b,o] = sum_q h21[b,q] * (wgt12[b,:] @ W_q[:,o]) + wgt12[b,:] @ B22[:,o]
//
// Wave tile = 32 rows x 32 cols (2 A row-tiles x 2 n-tiles).  The two A
// row-tiles (wgt12 fragments, 128 VGPRs) stay register-resident for all 256 q
// and SHARE each B fragment:  wmma(A0,b,..); wmma(A1,b,..)  -> 1 b128 load per
// WMMA and half the B bytes per MAC vs a 16-row tile.  Per q, each n-tile is
// processed sequentially with a zeroed C-temp chain (keeps only 16 VGPRs of
// temps live), then folded into the f32 accumulators scaled by h21[b,q]
// (staged per 32-q block in LDS).
// WG = 8 waves 2(M) x 4(N) -> tile 64 rows x 128 cols.  Grid (64, 2).
// ---------------------------------------------------------------------------
__global__ __launch_bounds__(256) void hyper_kernel(
        const __bf16* __restrict__ Aw12,   // [4096][256] bf16
        const float*  __restrict__ h21,    // [4096][256] f32
        const __bf16* __restrict__ Wp,     // [256][8][16][32][16] bf16 (packed w22)
        const __bf16* __restrict__ b22p,   // [8][16][32][16] bf16 (packed b22)
        float* __restrict__ outp) {        // [4096][256] f32
    __shared__ float hlds[2048];           // 64 rows x 32 q
    int lane = threadIdx.x & 31, g = lane >> 4, ln15 = lane & 15;
    int wave = threadIdx.x >> 5;
    int waveM = wave >> 2;                 // 0..1 : 32-row slab
    int waveN = wave & 3;                  // 0..3 : 32-col slab
    int m0 = blockIdx.x * 64;
    int r0 = m0 + waveM * 32;
    int c0 = blockIdx.y * 128 + waveN * 32;
    int nt0 = c0 >> 4;
    int rA0 = waveM * 32 + 8 * g;          // in-tile row of acc element j (row-tile 0)

    // A fragments: wgt12 rows r0..r0+31 (two 16-row tiles), all 8 K-chunks.
    const __bf16* arow0 = Aw12 + (long)(r0 + ln15) * 256;
    const __bf16* arow1 = Aw12 + (long)(r0 + 16 + ln15) * 256;
    bf16x16 wfrag[2][8];
    #pragma unroll
    for (int ch = 0; ch < 8; ++ch) {
        wfrag[0][ch] = load_afrag(arow0, ch * 32, g);
        wfrag[1][ch] = load_afrag(arow1, ch * 32, g);
    }

    const f32x8 zero = {0,0,0,0,0,0,0,0};
    f32x8 acc[2][2];                       // [row-tile][n-tile]
    acc[0][0] = zero; acc[0][1] = zero; acc[1][0] = zero; acc[1][1] = zero;

    for (int qb = 0; qb < 256; qb += 32) {
        __syncthreads();
        #pragma unroll
        for (int j = 0; j < 8; ++j) {      // stage h21[m0..m0+63, qb..qb+31]
            int lin = threadIdx.x + j * 256;
            hlds[lin] = h21[(long)(m0 + (lin >> 5)) * 256 + qb + (lin & 31)];
        }
        __syncthreads();

        #pragma unroll 1
        for (int kk = 0; kk < 32; ++kk) {
            int q = qb + kk;
            const __bf16* Wq = Wp + (long)q * 65536;
            float h0[8], h1[8];
            #pragma unroll
            for (int j = 0; j < 8; ++j) {
                h0[j] = hlds[(rA0 + j) * 32 + kk];
                h1[j] = hlds[(rA0 + 16 + j) * 32 + kk];
            }
            #pragma unroll
            for (int nt = 0; nt < 2; ++nt) {
                f32x8 ct0 = zero, ct1 = zero;
                #pragma unroll
                for (int ch = 0; ch < 8; ++ch) {
                    bf16x16 b = load_bfrag(Wq, (long)ch * 16 + nt0 + nt, lane);
                    ct0 = wmma_bf16(wfrag[0][ch], b, ct0);
                    ct1 = wmma_bf16(wfrag[1][ch], b, ct1);
                }
                #pragma unroll
                for (int j = 0; j < 8; ++j) {
                    acc[0][nt][j] += h0[j] * ct0[j];
                    acc[1][nt][j] += h1[j] * ct1[j];
                }
            }
            if (q < 255)  // pull next q's B stream toward L0/L2
                __builtin_prefetch(Wq + 65536 + (long)nt0 * 512 + lane * 16, 0, 1);
        }
    }
    // bias term: + wgt12 @ B22 (scale 1 -> accumulate directly)
    #pragma unroll
    for (int nt = 0; nt < 2; ++nt)
        #pragma unroll
        for (int ch = 0; ch < 8; ++ch) {
            bf16x16 b = load_bfrag(b22p, (long)ch * 16 + nt0 + nt, lane);
            acc[0][nt] = wmma_bf16(wfrag[0][ch], b, acc[0][nt]);
            acc[1][nt] = wmma_bf16(wfrag[1][ch], b, acc[1][nt]);
        }
    #pragma unroll
    for (int mi = 0; mi < 2; ++mi)
        #pragma unroll
        for (int nt = 0; nt < 2; ++nt)
            #pragma unroll
            for (int j = 0; j < 8; ++j) {
                int row = r0 + mi * 16 + 8 * g + j;
                outp[(long)row * 256 + c0 + nt * 16 + ln15] = acc[mi][nt][j];
            }
}

// ---------------------------------------------------------------------------
// LayerNorm(256) + ReLU, f32 in -> bf16 out. One wave per row, 8 rows per WG.
// ---------------------------------------------------------------------------
__global__ __launch_bounds__(256) void ln_relu_kernel(
        const float* __restrict__ inp, const float* __restrict__ gam,
        const float* __restrict__ bet, __bf16* __restrict__ outb) {
    int lane = threadIdx.x & 31;
    int row  = blockIdx.x * 8 + (threadIdx.x >> 5);
    float x[8], s = 0.f;
    #pragma unroll
    for (int j = 0; j < 8; ++j) { x[j] = inp[(long)row * 256 + lane + 32 * j]; s += x[j]; }
    float mean = wave_sum(s) * (1.f / 256.f);
    float vq = 0.f;
    #pragma unroll
    for (int j = 0; j < 8; ++j) { x[j] -= mean; vq += x[j] * x[j]; }
    float rs = rsqrtf(wave_sum(vq) * (1.f / 256.f) + 1e-5f);
    #pragma unroll
    for (int j = 0; j < 8; ++j) {
        int col = lane + 32 * j;
        float y = fmaxf(x[j] * rs * gam[col] + bet[col], 0.f);
        outb[(long)row * 256 + col] = (__bf16)y;
    }
}

// ---------------------------------------------------------------------------
// Final: Y = X @ w3 (bf16 WMMA, full N=256 per WG), then in-LDS row LayerNorm
// + ReLU -> f32 out.  WG = 8 waves 2(M)x4(N), tile 64 x 256.  Grid 64.
// ---------------------------------------------------------------------------
__global__ __launch_bounds__(256) void final_gemm_ln_kernel(
        const __bf16* __restrict__ X, const __bf16* __restrict__ w3p,
        const float* __restrict__ gam, const float* __restrict__ bet,
        float* __restrict__ out) {
    __shared__ float tile[64 * 256];
    int lane = threadIdx.x & 31, g = lane >> 4, ln15 = lane & 15;
    int wave = threadIdx.x >> 5;
    int waveM = wave >> 2, waveN = wave & 3;
    int m0 = blockIdx.x * 64;
    int r0 = m0 + waveM * 32;
    int c0 = waveN * 64;

    f32x8 acc[2][4];
    #pragma unroll
    for (int mi = 0; mi < 2; ++mi)
        #pragma unroll
        for (int t = 0; t < 4; ++t) acc[mi][t] = (f32x8){0,0,0,0,0,0,0,0};

    const __bf16* arow0 = X + (long)(r0 + ln15) * 256;
    const __bf16* arow1 = X + (long)(r0 + 16 + ln15) * 256;
    for (int ks = 0; ks < 8; ++ks) {
        bf16x16 a0 = load_afrag(arow0, ks * 32, g);
        bf16x16 a1 = load_afrag(arow1, ks * 32, g);
        #pragma unroll
        for (int t = 0; t < 4; ++t) {
            bf16x16 b = load_bfrag(w3p, (long)ks * 16 + (c0 >> 4) + t, lane);
            acc[0][t] = wmma_bf16(a0, b, acc[0][t]);
            acc[1][t] = wmma_bf16(a1, b, acc[1][t]);
        }
    }
    #pragma unroll
    for (int mi = 0; mi < 2; ++mi)
        #pragma unroll
        for (int t = 0; t < 4; ++t)
            #pragma unroll
            for (int j = 0; j < 8; ++j)
                tile[(waveM * 32 + mi * 16 + 8 * g + j) * 256 + c0 + t * 16 + ln15] =
                    acc[mi][t][j];
    __syncthreads();

    #pragma unroll 1
    for (int rr = 0; rr < 8; ++rr) {
        int row = wave * 8 + rr;
        float x[8], s = 0.f;
        #pragma unroll
        for (int j = 0; j < 8; ++j) { x[j] = tile[row * 256 + lane + 32 * j]; s += x[j]; }
        float mean = wave_sum(s) * (1.f / 256.f);
        float vq = 0.f;
        #pragma unroll
        for (int j = 0; j < 8; ++j) { x[j] -= mean; vq += x[j] * x[j]; }
        float rs = rsqrtf(wave_sum(vq) * (1.f / 256.f) + 1e-5f);
        #pragma unroll
        for (int j = 0; j < 8; ++j) {
            int col = lane + 32 * j;
            out[(long)(m0 + row) * 256 + col] =
                fmaxf(x[j] * rs * gam[col] + bet[col], 0.f);
        }
    }
}

__global__ void copy_kernel(const float* __restrict__ src, float* __restrict__ dst) {
    int idx = blockIdx.x * 256 + threadIdx.x;
    dst[idx] = src[idx];
}

// ---------------------------------------------------------------------------
// Workspace layout (bytes)
// ---------------------------------------------------------------------------
#define WS_CATB   0L          // 4096*512  bf16 = 4 MiB
#define WS_W11P   4194304L    // 512*256   bf16
#define WS_W21P   4456448L
#define WS_W12P   4718592L    // 256*256   bf16
#define WS_W3P    4849664L
#define WS_B22P   4980736L    // 256*256   bf16
#define WS_H11B   5111808L    // 4096*256  bf16
#define WS_WG12B  7208960L
#define WS_XLN    9306112L
#define WS_H21F   11403264L   // 4096*256  f32
#define WS_OUTP   15597568L
#define WS_W22P   19791872L   // 256*65536 bf16 = 32 MiB   (total ~50.9 MiB)

extern "C" void kernel_launch(void* const* d_in, const int* in_sizes, int n_in,
                              void* d_out, int out_size, void* d_ws, size_t ws_size,
                              hipStream_t stream) {
    (void)in_sizes; (void)n_in; (void)out_size; (void)ws_size;
    const float* img    = (const float*)d_in[0];
    const float* loc    = (const float*)d_in[1];
    const float* w11    = (const float*)d_in[2];
    const float* b11    = (const float*)d_in[3];
    const float* w12    = (const float*)d_in[4];
    const float* b12    = (const float*)d_in[5];
    const float* w21    = (const float*)d_in[6];
    const float* b21    = (const float*)d_in[7];
    const float* w22    = (const float*)d_in[8];
    const float* b22    = (const float*)d_in[9];
    const float* lnbr_g = (const float*)d_in[10];
    const float* lnbr_b = (const float*)d_in[11];
    const float* w3     = (const float*)d_in[12];
    const float* ln3_g  = (const float*)d_in[13];
    const float* ln3_b  = (const float*)d_in[14];

    char* ws = (char*)d_ws;
    __bf16* catB  = (__bf16*)(ws + WS_CATB);
    __bf16* w11p  = (__bf16*)(ws + WS_W11P);
    __bf16* w21p  = (__bf16*)(ws + WS_W21P);
    __bf16* w12p  = (__bf16*)(ws + WS_W12P);
    __bf16* w3p   = (__bf16*)(ws + WS_W3P);
    __bf16* b22p  = (__bf16*)(ws + WS_B22P);
    __bf16* h11B  = (__bf16*)(ws + WS_H11B);
    __bf16* wg12B = (__bf16*)(ws + WS_WG12B);
    __bf16* xln   = (__bf16*)(ws + WS_XLN);
    float*  h21f  = (float*) (ws + WS_H21F);
    float*  outp  = (float*) (ws + WS_OUTP);
    __bf16* w22p  = (__bf16*)(ws + WS_W22P);

    // ---- pack to bf16 (weights in fragment-linear WMMA-B layout) ----
    pack_cat_kernel<<<8192, 256, 0, stream>>>(img, loc, catB);
    pack_b_kernel<<<512,   256, 0, stream>>>(w11, w11p, 131072L, 131072L, 131072L);
    pack_b_kernel<<<512,   256, 0, stream>>>(w21, w21p, 131072L, 131072L, 131072L);
    pack_b_kernel<<<256,   256, 0, stream>>>(w12, w12p, 65536L, 65536L, 65536L);
    pack_b_kernel<<<256,   256, 0, stream>>>(w3,  w3p,  65536L, 65536L, 65536L);
    pack_b_kernel<<<256,   256, 0, stream>>>(b22, b22p, 65536L, 65536L, 65536L);
    pack_b_kernel<<<65536, 256, 0, stream>>>(w22, w22p, 65536L, 65536L, 65536L); // 256 batches

    // ---- front GEMMs ----
    dim3 g64x4(64, 4);
    gemm_bf16_kernel<<<g64x4, 256, 0, stream>>>(catB, 512, w11p, b11, h11B, 256, 16, 1, 1);
    gemm_bf16_kernel<<<g64x4, 256, 0, stream>>>(catB, 512, w21p, b21, h21f, 256, 16, 0, 1);
    gemm_bf16_kernel<<<g64x4, 256, 0, stream>>>(h11B, 256, w12p, b12, wg12B, 256, 8, 1, 0);

    // ---- fused hypernetwork bilinear (137 GFLOP) ----
    hyper_kernel<<<dim3(64, 2), 256, 0, stream>>>(wg12B, h21f, w22p, b22p, outp);

    // ---- epilogue: LN+ReLU, GEMM w3, LN+ReLU; second tuple element = loc_fea ----
    ln_relu_kernel<<<512, 256, 0, stream>>>(outp, lnbr_g, lnbr_b, xln);
    final_gemm_ln_kernel<<<64, 256, 0, stream>>>(xln, w3p, ln3_g, ln3_b, (float*)d_out);
    copy_kernel<<<4096, 256, 0, stream>>>(loc, (float*)d_out + (long)NROW * PL);
}
